// VisAttentionGNN_83021717832485
// MI455X (gfx1250) — compile-verified, matrix operand order checked by reading, not run on hardware
//
#include <hip/hip_runtime.h>
#include <math.h>

typedef __attribute__((ext_vector_type(16))) _Float16 v16h;
typedef __attribute__((ext_vector_type(8)))  float    v8f;

#define NEG_SLOPE 0.2f

// ---------------------------------------------------------------------------
// WMMA GEMM: H[n x oc] = A[n x 64] * W[64 x ldw]   (f16 operands, f32 accum)
// One wave per 16x16 output tile; K=64 -> two straight-line 16x16x32 WMMAs.
// Requires n % 16 == 0 (grid.x = n/16); W pre-padded to ldw columns so all
// loads are unconditional; only the store has a (uniform-ish) column guard.
// ---------------------------------------------------------------------------
__global__ __launch_bounds__(32) void gat_gemm_wmma(
    const float* __restrict__ A, const float* __restrict__ W,
    float* __restrict__ H, int n, int oc, int ldw) {
  const int tileM = blockIdx.x * 16;
  const int tileN = blockIdx.y * 16;
  const int lane  = threadIdx.x;          // 0..31 (wave32)
  const int mr    = lane & 15;
  const int kb    = (lane >> 4) << 3;     // K base: 0 (lanes 0-15) / 8 (16-31)

  const float* __restrict__ arow = A + (size_t)(tileM + mr) * 64;
  const int col = tileN + mr;             // always < ldw by construction

  v8f acc = {};
  #pragma unroll
  for (int kc = 0; kc < 64; kc += 32) {
    // A fragment: halves j=0..7 -> K=kc+kb+j ; j=8..15 -> K=kc+kb+16+(j-8)
    const float4 a0 = *(const float4*)(arow + kc + kb);
    const float4 a1 = *(const float4*)(arow + kc + kb + 4);
    const float4 a2 = *(const float4*)(arow + kc + kb + 16);
    const float4 a3 = *(const float4*)(arow + kc + kb + 20);
    v16h a, b;
    a[0]  = (_Float16)a0.x; a[1]  = (_Float16)a0.y;
    a[2]  = (_Float16)a0.z; a[3]  = (_Float16)a0.w;
    a[4]  = (_Float16)a1.x; a[5]  = (_Float16)a1.y;
    a[6]  = (_Float16)a1.z; a[7]  = (_Float16)a1.w;
    a[8]  = (_Float16)a2.x; a[9]  = (_Float16)a2.y;
    a[10] = (_Float16)a2.z; a[11] = (_Float16)a2.w;
    a[12] = (_Float16)a3.x; a[13] = (_Float16)a3.y;
    a[14] = (_Float16)a3.z; a[15] = (_Float16)a3.w;
    const int wbase = (kc + kb) * ldw + col;   // 32-bit offsets, strength-reduced
    #pragma unroll
    for (int j = 0; j < 8; ++j) {
      b[j]     = (_Float16)W[wbase + j * ldw];
      b[j + 8] = (_Float16)W[wbase + (16 + j) * ldw];
    }
    acc = __builtin_amdgcn_wmma_f32_16x16x32_f16(
        /*neg_a=*/false, a, /*neg_b=*/false, b,
        /*c_mod=*/(short)0, acc, /*reuse_a=*/false, /*reuse_b=*/false);
  }
  // C/D layout: VGPR r holds M = rowBase + r; rows in range by construction.
  const int rowBase = tileM + ((lane >> 4) << 3);
  if (col < oc) {
    float* __restrict__ p = H + (size_t)rowBase * oc + col;
    #pragma unroll
    for (int r = 0; r < 8; ++r) p[r * oc] = acc[r];
  }
}

// Zero-pad W [kdim x oc] -> Wp [kdim x ocp]
__global__ void gat_pad_w(const float* __restrict__ W, float* __restrict__ Wp,
                          int kdim, int oc, int ocp) {
  int i = blockIdx.x * blockDim.x + threadIdx.x;
  if (i >= kdim * ocp) return;
  int k = i / ocp, c = i % ocp;
  Wp[i] = (c < oc) ? W[k * oc + c] : 0.f;
}

// ---------------------------------------------------------------------------
// Self-loop attr: cnt[d] += 1, loop_attr[d] += edge_attr[e]  (once per graph)
// ---------------------------------------------------------------------------
__global__ void gat_loopattr_acc(const int* __restrict__ dst,
                                 const float* __restrict__ edge_attr,
                                 float* cnt, float* loop_attr, int E_) {
  int i = blockIdx.x * blockDim.x + threadIdx.x;
  if (i >= E_) return;
  int d = dst[i];
  atomicAdd(&cnt[d], 1.0f);
  #pragma unroll
  for (int t = 0; t < 8; ++t)
    atomicAdd(&loop_attr[(size_t)d * 8 + t], edge_attr[(size_t)i * 8 + t]);
}

__global__ void gat_loopattr_div(float* loop_attr, const float* cnt, int n) {
  int i = blockIdx.x * blockDim.x + threadIdx.x;
  if (i >= n * 8) return;
  float c = cnt[i >> 3];
  loop_attr[i] /= (c < 1.f ? 1.f : c);
}

// ---------------------------------------------------------------------------
// Per-node attention logits: al_src/al_dst[n,h] = <h[n,h,:], a_src/dst[h,:]>
// ---------------------------------------------------------------------------
__global__ void gat_node_al(const float* __restrict__ Hf,
                            const float* __restrict__ a_src,
                            const float* __restrict__ a_dst,
                            float* al_src, float* al_dst,
                            int n, int H_, int C_) {
  int idx = blockIdx.x * blockDim.x + threadIdx.x;
  if (idx >= n * H_) return;
  int node = idx / H_, h = idx % H_;
  const float* row = Hf + (size_t)node * H_ * C_ + (size_t)h * C_;
  float s = 0.f, d = 0.f;
  for (int c = 0; c < C_; ++c) {
    float v = row[c];
    s += v * a_src[h * C_ + c];
    d += v * a_dst[h * C_ + c];
  }
  al_src[idx] = s;
  al_dst[idx] = d;
}

// q_e[t,h] = sum_c We[t, h*C+c] * a_e[h,c]   (8*H values)
__global__ void gat_qe(const float* __restrict__ We, const float* __restrict__ ae,
                       float* qe, int H_, int C_) {
  int idx = threadIdx.x;
  if (idx >= 8 * H_) return;
  int t = idx / H_, h = idx % H_;
  float s = 0.f;
  for (int c = 0; c < C_; ++c)
    s += We[(size_t)t * H_ * C_ + h * C_ + c] * ae[h * C_ + c];
  qe[t * H_ + h] = s;
}

// ---------------------------------------------------------------------------
// Edge pass 1: alpha = leakyrelu(al_src[s]+al_dst[d]+<ea,q>); atomicMax amax[d]
// Monotone float->uint key so unsigned atomicMax implements float max.
// ---------------------------------------------------------------------------
__device__ __forceinline__ unsigned fkey(float f) {
  unsigned b = __float_as_uint(f);
  return (b & 0x80000000u) ? ~b : (b | 0x80000000u);
}
__device__ __forceinline__ float fkey_inv(unsigned k) {
  unsigned b = (k & 0x80000000u) ? (k & 0x7FFFFFFFu) : ~k;
  return __uint_as_float(b);
}

__global__ void gat_edge_alpha(const int* __restrict__ src_idx,
                               const int* __restrict__ dst_idx,
                               const float* __restrict__ edge_attr,
                               const float* __restrict__ loop_attr,
                               const float* __restrict__ al_src,
                               const float* __restrict__ al_dst,
                               const float* __restrict__ qe,
                               float* alpha, unsigned* amax,
                               int E_, int n, int H_) {
  int i = blockIdx.x * blockDim.x + threadIdx.x;
  if (i >= E_ + n) return;
  int s, d; const float* ea;
  if (i < E_) { s = src_idx[i]; d = dst_idx[i]; ea = edge_attr + (size_t)i * 8; }
  else        { s = d = i - E_;                 ea = loop_attr + (size_t)(i - E_) * 8; }
  float eav[8];
  #pragma unroll
  for (int t = 0; t < 8; ++t) eav[t] = ea[t];
  for (int h = 0; h < H_; ++h) {
    float r = al_src[s * H_ + h] + al_dst[d * H_ + h];
    #pragma unroll
    for (int t = 0; t < 8; ++t) r += eav[t] * qe[t * H_ + h];
    float a = (r > 0.f) ? r : NEG_SLOPE * r;
    alpha[(size_t)i * H_ + h] = a;
    atomicMax(&amax[d * H_ + h], fkey(a));
  }
}

// ---------------------------------------------------------------------------
// Edge pass 2: ex = exp(alpha - amax[d]); denom[d]+=ex; num[d,h,:]+=ex*h[s,h,:]
// ---------------------------------------------------------------------------
__global__ void gat_edge_accum(const int* __restrict__ src_idx,
                               const int* __restrict__ dst_idx,
                               const float* __restrict__ alpha,
                               const unsigned* __restrict__ amax,
                               const float* __restrict__ Hf,
                               float* denom, float* num,
                               int E_, int n, int H_, int C_) {
  long idx = (long)blockIdx.x * blockDim.x + threadIdx.x;
  if (idx >= (long)(E_ + n) * H_) return;
  int i = (int)(idx / H_), h = (int)(idx % H_);
  int s, d;
  if (i < E_) { s = src_idx[i]; d = dst_idx[i]; }
  else        { s = d = i - E_; }
  float mx = fkey_inv(amax[d * H_ + h]);
  float ex = __expf(alpha[(size_t)i * H_ + h] - mx);
  atomicAdd(&denom[d * H_ + h], ex);
  const float* hrow = Hf + (size_t)s * H_ * C_ + (size_t)h * C_;
  float* nrow       = num + (size_t)d * H_ * C_ + (size_t)h * C_;
  for (int c = 0; c < C_; ++c) atomicAdd(&nrow[c], ex * hrow[c]);
}

// ---------------------------------------------------------------------------
// Node epilogue: out = num/denom (concat or mean over heads) + bias, opt. relu
// ---------------------------------------------------------------------------
__global__ void gat_node_final(const float* __restrict__ num,
                               const float* __restrict__ denom,
                               const float* __restrict__ bias,
                               float* out, int n, int H_, int C_,
                               int concat, int do_relu) {
  int OC = concat ? H_ * C_ : C_;
  long idx = (long)blockIdx.x * blockDim.x + threadIdx.x;
  if (idx >= (long)n * OC) return;
  int node = (int)(idx / OC), c = (int)(idx % OC);
  float v;
  if (concat) {
    int h = c / C_;
    v = num[(size_t)node * H_ * C_ + c] / (denom[node * H_ + h] + 1e-16f);
  } else {
    v = 0.f;
    for (int h = 0; h < H_; ++h)
      v += num[(size_t)node * H_ * C_ + h * C_ + c] / (denom[node * H_ + h] + 1e-16f);
    v /= (float)H_;
  }
  v += bias[c];
  if (do_relu) v = (v > 0.f) ? v : 0.f;
  out[(size_t)node * OC + c] = v;
}

// ---------------------------------------------------------------------------
extern "C" void kernel_launch(void* const* d_in, const int* in_sizes, int n_in,
                              void* d_out, int out_size, void* d_ws, size_t ws_size,
                              hipStream_t stream) {
  const float* x         = (const float*)d_in[0];
  const int*   edge_idx  = (const int*)d_in[1];
  const float* edge_attr = (const float*)d_in[2];

  const int N  = in_sizes[0] / 64;
  const int E  = in_sizes[1] / 2;
  const int EN = E + N;
  const int* src = edge_idx;
  const int* dst = edge_idx + E;

  // Per-layer parameter tables: W, a_src, a_dst, We, a_e, b at d_in[3+6*l ...]
  const float* Wl[4];  const float* asl[4]; const float* adl[4];
  const float* Wel[4]; const float* ael[4]; const float* bl[4];
  for (int l = 0; l < 4; ++l) {
    Wl[l]  = (const float*)d_in[3 + 6 * l + 0];
    asl[l] = (const float*)d_in[3 + 6 * l + 1];
    adl[l] = (const float*)d_in[3 + 6 * l + 2];
    Wel[l] = (const float*)d_in[3 + 6 * l + 3];
    ael[l] = (const float*)d_in[3 + 6 * l + 4];
    bl[l]  = (const float*)d_in[3 + 6 * l + 5];
  }
  const int  Hh[4]     = {1, 2, 2, 1};
  const int  Cc[4]     = {64, 64, 64, 2};
  const int  concat[4] = {1, 0, 0, 1};
  const int  relu[4]   = {1, 1, 1, 0};

  // Workspace carve-up
  char* ws = (char*)d_ws;
  size_t off = 0;
  auto carve = [&](size_t bytes) { void* p = ws + off; off += (bytes + 255) & ~255ull; return p; };
  float*    act0      = (float*)carve((size_t)N * 64 * 4);
  float*    act1      = (float*)carve((size_t)N * 64 * 4);
  float*    hbuf      = (float*)carve((size_t)N * 128 * 4);
  float*    numb      = (float*)carve((size_t)N * 128 * 4);
  float*    alpha     = (float*)carve((size_t)EN * 2 * 4);
  unsigned* amax      = (unsigned*)carve((size_t)N * 2 * 4);
  float*    denom     = (float*)carve((size_t)N * 2 * 4);
  float*    al_src    = (float*)carve((size_t)N * 2 * 4);
  float*    al_dst    = (float*)carve((size_t)N * 2 * 4);
  float*    cnt       = (float*)carve((size_t)N * 4);
  float*    loop_attr = (float*)carve((size_t)N * 8 * 4);
  float*    qe        = (float*)carve(64 * 4);
  float*    w3pad     = (float*)carve((size_t)64 * 16 * 4);
  (void)ws_size;

  const int TB = 256;

  // Graph-constant self-loop attrs (scatter-mean of edge_attr over dst)
  hipMemsetAsync(cnt, 0, (size_t)N * 4, stream);
  hipMemsetAsync(loop_attr, 0, (size_t)N * 8 * 4, stream);
  gat_loopattr_acc<<<(E + TB - 1) / TB, TB, 0, stream>>>(dst, edge_attr, cnt, loop_attr, E);
  gat_loopattr_div<<<(N * 8 + TB - 1) / TB, TB, 0, stream>>>(loop_attr, cnt, N);

  // Pad the final 64x2 projection to 64x16 so GEMM loads are unconditional
  gat_pad_w<<<(64 * 16 + TB - 1) / TB, TB, 0, stream>>>(Wl[3], w3pad, 64, 2, 16);

  const float* act_in = x;
  for (int l = 0; l < 4; ++l) {
    const int H_ = Hh[l], C_ = Cc[l], HC = H_ * C_;
    const int OC = concat[l] ? HC : C_;
    const int ldw = (HC + 15) & ~15;            // padded GEMM width
    const float* Wg = (l == 3) ? w3pad : Wl[l];
    float* act_out = (l == 0) ? act0 : (l == 1) ? act1 : (l == 2) ? act0 : (float*)d_out;

    hipMemsetAsync(amax,  0, (size_t)N * H_ * 4, stream);
    hipMemsetAsync(denom, 0, (size_t)N * H_ * 4, stream);
    hipMemsetAsync(numb,  0, (size_t)N * HC * 4, stream);

    // h = act_in @ W   (WMMA); N assumed multiple of 16 (50000 = 3125*16)
    dim3 gg(N / 16, ldw / 16);
    gat_gemm_wmma<<<gg, 32, 0, stream>>>(act_in, Wg, hbuf, N, HC, ldw);

    gat_node_al<<<(N * H_ + TB - 1) / TB, TB, 0, stream>>>(
        hbuf, asl[l], adl[l], al_src, al_dst, N, H_, C_);
    gat_qe<<<1, 16, 0, stream>>>(Wel[l], ael[l], qe, H_, C_);

    gat_edge_alpha<<<(EN + TB - 1) / TB, TB, 0, stream>>>(
        src, dst, edge_attr, loop_attr, al_src, al_dst, qe, alpha, amax, E, N, H_);

    long nacc = (long)EN * H_;
    gat_edge_accum<<<(unsigned)((nacc + TB - 1) / TB), TB, 0, stream>>>(
        src, dst, alpha, amax, hbuf, denom, numb, E, N, H_, C_);

    long nfin = (long)N * OC;
    gat_node_final<<<(unsigned)((nfin + TB - 1) / TB), TB, 0, stream>>>(
        numb, denom, bl[l], act_out, N, H_, C_, concat[l], relu[l]);

    act_in = act_out;
  }
  (void)out_size; (void)n_in;
}
